// TsLstm3_34471407517910
// MI455X (gfx1250) — matrix operand order repaired and checked by reading, not compiled, not run on hardware
//
#include <hip/hip_runtime.h>

// ---------------------------------------------------------------------------
// Persistent 3-layer LSTM for MI455X (gfx1250, wave32, WMMA).
//   grid = 64 WGs x 256 thr (8 waves). WG owns batch rows [16*wg, 16*wg+16).
//   Per step: GEMM1 (h1*Whh1^T), cell1 -> GEMM2 (c1*Wih2^T + h2*Whh2^T),
//   cell2 -> tiny layer3 -> next step. States stay in regs/LDS for 576 steps.
//   Weights pre-converted to bf16 WMMA B-fragment layout in d_ws (prep kernel).
//   Weight bases are opaque per-iteration (defeats LICM -> no spilling) AND
//   explicitly address_space(1) so loads lower to global_load_b128 (LOADcnt
//   only, saddr + immediate offsets) instead of flat_load (LOADcnt+DScnt).
// ---------------------------------------------------------------------------

typedef __attribute__((ext_vector_type(16))) __bf16 v16bf;
typedef __attribute__((ext_vector_type(8)))  float  v8f;
typedef const __attribute__((address_space(1))) v16bf* gvp;   // global-AS fragment ptr

#define T_STEPS 512
#define FUT     64
#define BATCH   1024
#define HID     256
#define W_ELEMS (64*8*32*16)   // 262144 bf16 per weight matrix (ntile,ktile,lane,elem)

__device__ __forceinline__ float sigm(float x) { return 1.0f / (1.0f + __expf(-x)); }

__device__ __forceinline__ v8f zero8() {
  v8f z;
  for (int j = 0; j < 8; ++j) z[j] = 0.0f;
  return z;
}

// One GEMM pass: acc[idx] += A(16x256 in LDS frags) * B(256 x 8 tiles).
// B loads are rotated one tile ahead of the consuming WMMA.
__device__ __forceinline__ void gemm_pass(const v16bf* A_lds, gvp B,
                                          int w, int lane, v8f acc[8]) {
#pragma unroll
  for (int kt = 0; kt < 8; ++kt) {
    v16bf a = A_lds[kt * 32 + lane];
    v16bf b = B[((2 * w) * 8 + kt) * 32 + lane];   // idx=0 tile: nt = 2w
#pragma unroll
    for (int idx = 0; idx < 8; ++idx) {
      v16bf bn = b;
      if (idx < 7) {
        int g = (idx + 1) >> 1, tt = (idx + 1) & 1;
        int nt = 16 * g + 2 * w + tt;
        bn = B[(nt * 8 + kt) * 32 + lane];          // prefetch next tile
      }
      acc[idx] = __builtin_amdgcn_wmma_f32_16x16x32_bf16(
          false, a, false, b, (short)0, acc[idx], false, false);
      b = bn;
    }
  }
}

// --- prep: scatter f32 weights into bf16 WMMA B-fragment layout -------------
// B (32x16 bf16): lane l -> n = l&15, hi = l>>4 ; element i -> k = 16*(i>>3) + 8*hi + (i&7).
__global__ void prep_weights(const float* __restrict__ whh1,
                             const float* __restrict__ wih2,
                             const float* __restrict__ whh2,
                             __bf16* __restrict__ wsbf) {
  int idx = blockIdx.x * blockDim.x + threadIdx.x;
  if (idx >= 3 * W_ELEMS) return;
  int widx = idx / W_ELEMS;
  int rem  = idx % W_ELEMS;
  int i    = rem & 15;
  int frag = rem >> 4;
  int lane = frag & 31;
  int ntkt = frag >> 5;
  int kt   = ntkt & 7;
  int nt   = ntkt >> 3;
  int n    = nt * 16 + (lane & 15);
  int k    = kt * 32 + (i >> 3) * 16 + (lane >> 4) * 8 + (i & 7);
  const float* W = (widx == 0) ? whh1 : (widx == 1) ? wih2 : whh2;
  wsbf[idx] = (__bf16)W[n * HID + k];
}

__global__ __launch_bounds__(256)
void lstm_persistent(const float* __restrict__ x,
                     const float* __restrict__ wih1,   // [1024] (4H x 1)
                     const float* __restrict__ bih1, const float* __restrict__ bhh1,
                     const float* __restrict__ bih2, const float* __restrict__ bhh2,
                     const float* __restrict__ wih3,   // [4,256]
                     const float* __restrict__ whh3,   // [4]
                     const float* __restrict__ bih3, const float* __restrict__ bhh3,
                     const __bf16* __restrict__ wf,    // 3 * W_ELEMS bf16 fragments
                     float* __restrict__ out)          // [1024, 64]
{
  __shared__ __align__(32) __bf16 sAh1[8 * 512];   // h1 A-fragments (bf16)
  __shared__ __align__(32) __bf16 sAc1[8 * 512];   // c1 A-fragments
  __shared__ __align__(32) __bf16 sAh2[8 * 512];   // h2 A-fragments
  __shared__ float sC2[16 * 256];                  // c2 plain f32 (layer-3 input)
  __shared__ float sWih3[4 * 256];
  __shared__ float sXin[16];                       // layer-1 scalar input per row
  __shared__ float sH3[16], sC3[16];
  __shared__ float sZ3[64];
  __shared__ float sB3[4], sWhh3[4];

  const int tid   = threadIdx.x;
  const int lane  = tid & 31;
  const int w     = tid >> 5;       // wave id 0..7 == K-tile it owns
  const int nl    = lane & 15;
  const int hi    = lane >> 4;
  const int bbase = blockIdx.x * 16;

  // ---- init on-chip state ----
  for (int i = tid; i < 8 * 512; i += 256) {
    sAh1[i] = (__bf16)0.0f; sAc1[i] = (__bf16)0.0f; sAh2[i] = (__bf16)0.0f;
  }
  for (int i = tid; i < 1024; i += 256) sWih3[i] = wih3[i];
  if (tid < 16) { sH3[tid] = 0.0f; sC3[tid] = 0.0f; }
  if (tid < 4)  { sB3[tid] = bih3[tid] + bhh3[tid]; sWhh3[tid] = whh3[tid]; }

  // ---- loop-invariant per-lane scalars: tile idx = g*2+tt, n = (16g+2w+tt)*16+nl
  float b1v[8], w1v[8], b2v[8];
#pragma unroll
  for (int idx = 0; idx < 8; ++idx) {
    int g = idx >> 1, tt = idx & 1;
    int n = (16 * g + 2 * w + tt) * 16 + nl;
    b1v[idx] = bih1[n] + bhh1[n];
    w1v[idx] = wih1[n];
    b2v[idx] = bih2[n] + bhh2[n];
  }

  float c1s[16], c2s[16];                // c-state: index tt*8+r <-> (m=r+8hi, hcol=32w+16tt+nl)
#pragma unroll
  for (int i = 0; i < 16; ++i) { c1s[i] = 0.0f; c2s[i] = 0.0f; }

  const v16bf* Ah1 = (const v16bf*)sAh1;
  const v16bf* Ac1 = (const v16bf*)sAc1;
  const v16bf* Ah2 = (const v16bf*)sAh2;

  __syncthreads();

  for (int t = 0; t < T_STEPS + FUT; ++t) {
    // Opaque, global-AS copies of the weight bases: the asm defeats LICM (no
    // hoist/spill across the time loop) and the explicit address_space(1)
    // keeps the loads lowering to global_load_b128 (LOADcnt only).
    gvp pWhh1 = (gvp)(wf);
    gvp pWih2 = (gvp)(wf + W_ELEMS);
    gvp pWhh2 = (gvp)(wf + 2 * W_ELEMS);
    asm volatile("" : "+s"(pWhh1), "+s"(pWih2), "+s"(pWhh2));

    if (t < T_STEPS && tid < 16) sXin[tid] = x[t * BATCH + bbase + tid];
    __syncthreads();

    // ================= GEMM1: z1 = h1 @ Whh1^T =================
    v8f acc[8];
#pragma unroll
    for (int idx = 0; idx < 8; ++idx) acc[idx] = zero8();
    gemm_pass(Ah1, pWhh1, w, lane, acc);
    __syncthreads();   // protect A_h1/A_c1 writes below vs other waves' reads

    // ================= Layer-1 cell update =====================
    {
      float xr[8];
#pragma unroll
      for (int r = 0; r < 8; ++r) xr[r] = sXin[r + 8 * hi];
#pragma unroll
      for (int tt = 0; tt < 2; ++tt) {
#pragma unroll
        for (int r = 0; r < 8; ++r) {
          float xm = xr[r];
          float zi = acc[0 + tt][r] + b1v[0 + tt] + xm * w1v[0 + tt];
          float zf = acc[2 + tt][r] + b1v[2 + tt] + xm * w1v[2 + tt];
          float zg = acc[4 + tt][r] + b1v[4 + tt] + xm * w1v[4 + tt];
          float zo = acc[6 + tt][r] + b1v[6 + tt] + xm * w1v[6 + tt];
          float cn = sigm(zf) * c1s[tt * 8 + r] + sigm(zi) * tanhf(zg);
          float hn = sigm(zo) * tanhf(cn);
          c1s[tt * 8 + r] = cn;
          // scatter into A-fragment layout: k_local = 16*tt + nl
          int la = r + 8 * hi + 16 * ((nl >> 3) & 1);
          int ie = (nl & 7) + 8 * tt;
          sAh1[w * 512 + la * 16 + ie] = (__bf16)hn;
          sAc1[w * 512 + la * 16 + ie] = (__bf16)cn;
        }
      }
    }
    __syncthreads();

    // ========== GEMM2: z2 = c1 @ Wih2^T + h2 @ Whh2^T ==========
#pragma unroll
    for (int idx = 0; idx < 8; ++idx) acc[idx] = zero8();
    gemm_pass(Ac1, pWih2, w, lane, acc);
    gemm_pass(Ah2, pWhh2, w, lane, acc);
    __syncthreads();   // protect A_h2 writes below vs other waves' reads

    // ================= Layer-2 cell update =====================
#pragma unroll
    for (int tt = 0; tt < 2; ++tt) {
#pragma unroll
      for (int r = 0; r < 8; ++r) {
        float zi = acc[0 + tt][r] + b2v[0 + tt];
        float zf = acc[2 + tt][r] + b2v[2 + tt];
        float zg = acc[4 + tt][r] + b2v[4 + tt];
        float zo = acc[6 + tt][r] + b2v[6 + tt];
        float cn = sigm(zf) * c2s[tt * 8 + r] + sigm(zi) * tanhf(zg);
        float hn = sigm(zo) * tanhf(cn);
        c2s[tt * 8 + r] = cn;
        int la = r + 8 * hi + 16 * ((nl >> 3) & 1);
        int ie = (nl & 7) + 8 * tt;
        sAh2[w * 512 + la * 16 + ie] = (__bf16)hn;
        int m = r + 8 * hi;
        sC2[m * 256 + 32 * w + 16 * tt + nl] = cn;   // plain f32 for layer 3
      }
    }
    __syncthreads();

    // ================= Layer 3 (H=1): z3[m][g] ==================
    {
      int p = tid >> 2;            // 64 (m,g) pairs
      int m = p >> 2;
      int g = p & 3;
      int sub = tid & 3;
      float partial = 0.0f;
#pragma unroll 8
      for (int j = 0; j < 64; ++j) {
        int k = sub + 4 * j;
        partial += sC2[m * 256 + k] * sWih3[g * 256 + k];
      }
      partial += __shfl_xor(partial, 1, 32);
      partial += __shfl_xor(partial, 2, 32);
      if (sub == 0) sZ3[m * 4 + g] = partial + sH3[m] * sWhh3[g] + sB3[g];
    }
    __syncthreads();
    if (tid < 16) {
      int m = tid;
      float zi = sZ3[4 * m + 0], zf = sZ3[4 * m + 1];
      float zg = sZ3[4 * m + 2], zo = sZ3[4 * m + 3];
      float cn = sigm(zf) * sC3[m] + sigm(zi) * tanhf(zg);
      float hn = sigm(zo) * tanhf(cn);
      sC3[m] = cn; sH3[m] = hn;
      sXin[m] = cn;                                   // next step's layer-1 input
      if (t >= T_STEPS) out[(bbase + m) * FUT + (t - T_STEPS)] = cn;
    }
    __syncthreads();
  }
}

extern "C" void kernel_launch(void* const* d_in, const int* in_sizes, int n_in,
                              void* d_out, int out_size, void* d_ws, size_t ws_size,
                              hipStream_t stream) {
  const float* x    = (const float*)d_in[0];
  const float* wih1 = (const float*)d_in[1];
  const float* whh1 = (const float*)d_in[2];
  const float* bih1 = (const float*)d_in[3];
  const float* bhh1 = (const float*)d_in[4];
  const float* wih2 = (const float*)d_in[5];
  const float* whh2 = (const float*)d_in[6];
  const float* bih2 = (const float*)d_in[7];
  const float* bhh2 = (const float*)d_in[8];
  const float* wih3 = (const float*)d_in[9];
  const float* whh3 = (const float*)d_in[10];
  const float* bih3 = (const float*)d_in[11];
  const float* bhh3 = (const float*)d_in[12];
  __bf16* wsbf = (__bf16*)d_ws;            // 3 * 262144 bf16 = 1.5 MB
  float* out = (float*)d_out;

  int total = 3 * W_ELEMS;
  prep_weights<<<(total + 255) / 256, 256, 0, stream>>>(whh1, wih2, whh2, wsbf);
  lstm_persistent<<<64, 256, 0, stream>>>(x, wih1, bih1, bhh1, bih2, bhh2,
                                          wih3, whh3, bih3, bhh3, wsbf, out);
}
